// Pts_3DGC_79147657331275
// MI455X (gfx1250) — matrix operand order repaired and checked by reading, not compiled, not run on hardware
//
#include <hip/hip_runtime.h>
#include <math.h>

typedef __attribute__((ext_vector_type(16))) _Float16 v16h;
typedef __attribute__((ext_vector_type(8)))  _Float16 v8h;
typedef __attribute__((ext_vector_type(4)))  _Float16 v4h;
typedef __attribute__((ext_vector_type(8)))  float    v8f;
typedef __attribute__((ext_vector_type(4)))  float    v4f;

#define NB 4        // batches
#define SUP 7       // SUPPORT_NUM
#define KNN_MAX 10

// ---------------------------------------------------------------------------
// Brute-force KNN: one thread per query point, insertion sort of K nearest.
// verts laid out (NB, vstride, 3); candidates are the first Vcand points.
// ---------------------------------------------------------------------------
__global__ void knn_kernel(const float* __restrict__ verts, int vstride,
                           int Vcand, int Vq, int K, int* __restrict__ idx)
{
    int t = blockIdx.x * blockDim.x + threadIdx.x;
    if (t >= NB * Vq) return;
    int b = t / Vq, q = t % Vq;
    const float* base = verts + (size_t)b * vstride * 3;
    float qx = base[q*3+0], qy = base[q*3+1], qz = base[q*3+2];
    float bd[KNN_MAX]; int bi[KNN_MAX];
    for (int i = 0; i < K; ++i){ bd[i] = 3.4e38f; bi[i] = 0; }
    for (int j = 0; j < Vcand; ++j){
        if (j == q) continue;
        float dx = base[j*3+0]-qx, dy = base[j*3+1]-qy, dz = base[j*3+2]-qz;
        float d = dx*dx + dy*dy + dz*dz;
        if (d < bd[K-1]){
            int p = K-1;
            while (p > 0 && bd[p-1] > d){ bd[p] = bd[p-1]; bi[p] = bi[p-1]; --p; }
            bd[p] = d; bi[p] = j;
        }
    }
    int* out = idx + (size_t)t * K;
    for (int i = 0; i < K; ++i) out[i] = bi[i];
}

// ---------------------------------------------------------------------------
// Column-normalize the direction matrix dirs (3 x J, row-major).
// ---------------------------------------------------------------------------
__global__ void normdirs_kernel(const float* __restrict__ d, int J,
                                float* __restrict__ nd)
{
    int j = blockIdx.x * blockDim.x + threadIdx.x;
    if (j >= J) return;
    float a = d[j], b = d[J + j], c = d[2*J + j];
    float inv = 1.f / fmaxf(sqrtf(a*a + b*b + c*c), 1e-12f);
    nd[j] = a*inv; nd[J + j] = b*inv; nd[2*J + j] = c*inv;
}

// ---------------------------------------------------------------------------
// Surface layer fused: out[b,v,c] = sum_s max_n relu(dnorm[n].dir[s*128+c])
//                                   + verts[b,v,:].ste[c,:]
// block = (v, b), 128 threads (one per channel).
// ---------------------------------------------------------------------------
__global__ void surface_kernel(const float* __restrict__ verts, int vstride,
                               const int* __restrict__ idx, int NN,
                               const float* __restrict__ nd, int J,
                               const float* __restrict__ ste,
                               float* __restrict__ out, int V)
{
    int v = blockIdx.x, b = blockIdx.y;
    int tid = threadIdx.x;
    __shared__ float sdn[KNN_MAX][3];
    const float* base = verts + (size_t)b * vstride * 3;
    float cx = base[v*3+0], cy = base[v*3+1], cz = base[v*3+2];
    if (tid < NN){
        int j = idx[((size_t)b*V + v)*NN + tid];
        float dx = base[j*3+0]-cx, dy = base[j*3+1]-cy, dz = base[j*3+2]-cz;
        float inv = 1.f / fmaxf(sqrtf(dx*dx + dy*dy + dz*dz), 1e-12f);
        sdn[tid][0] = dx*inv; sdn[tid][1] = dy*inv; sdn[tid][2] = dz*inv;
    }
    __syncthreads();
    int c = tid;  // C == 128 == blockDim.x
    float acc = 0.f;
    for (int s = 0; s < SUP; ++s){
        int jc = s*128 + c;
        float d0 = nd[jc], d1 = nd[J + jc], d2 = nd[2*J + jc];
        float m = 0.f;  // relu values are >= 0
        #pragma unroll 1
        for (int n = 0; n < NN; ++n){
            float th = fmaxf(sdn[n][0]*d0 + sdn[n][1]*d1 + sdn[n][2]*d2, 0.f);
            m = fmaxf(m, th);
        }
        acc += m;
    }
    float fste = cx*ste[c*3+0] + cy*ste[c*3+1] + cz*ste[c*3+2];
    out[((size_t)b*V + v)*128 + c] = acc + fste;
}

// ---------------------------------------------------------------------------
// HS layer fused activation:
// pre[b,v,c] = fout[b,v,c]                    (f_center)
//            + sum_s max_n relu(dnorm[n].dir[s*C+c]) * fout[b,idx[n],C+s*C+c]
//            + fste[b,v,c]
// block = (v, b), blockDim = min(C,256) threads looping channels.
// ---------------------------------------------------------------------------
__global__ void hs_act_kernel(const float* __restrict__ verts, int vstride,
                              const int* __restrict__ idx, int NN,
                              const float* __restrict__ nd, int J,
                              const float* __restrict__ fout, int ldo,
                              const float* __restrict__ fste,
                              float* __restrict__ pre, int V, int C)
{
    int v = blockIdx.x, b = blockIdx.y;
    int tid = threadIdx.x;
    __shared__ float  sdn[KNN_MAX][3];
    __shared__ size_t soff[KNN_MAX];  // base offset of each neighbor's support row
    if (tid < NN){
        const float* base = verts + (size_t)b * vstride * 3;
        float cx = base[v*3+0], cy = base[v*3+1], cz = base[v*3+2];
        int j = idx[((size_t)b*V + v)*NN + tid];
        soff[tid] = ((size_t)b*V + j)*ldo + C;
        float dx = base[j*3+0]-cx, dy = base[j*3+1]-cy, dz = base[j*3+2]-cz;
        float inv = 1.f / fmaxf(sqrtf(dx*dx + dy*dy + dz*dz), 1e-12f);
        sdn[tid][0] = dx*inv; sdn[tid][1] = dy*inv; sdn[tid][2] = dz*inv;
    }
    __syncthreads();
    size_t row = (size_t)b*V + v;
    const float* frow = fout + row*ldo;
    for (int c = tid; c < C; c += blockDim.x){
        float acc = 0.f;
        for (int s = 0; s < SUP; ++s){
            int jc = s*C + c;
            float d0 = nd[jc], d1 = nd[J + jc], d2 = nd[2*J + jc];
            float m = -3.4e38f;
            #pragma unroll 1
            for (int n = 0; n < NN; ++n){
                float th = fmaxf(sdn[n][0]*d0 + sdn[n][1]*d1 + sdn[n][2]*d2, 0.f);
                float fs = fout[soff[n] + jc];
                m = fmaxf(m, th*fs);
            }
            acc += m;
        }
        pre[row*C + c] = frow[c] + acc + fste[row*C + c];
    }
}

// ---------------------------------------------------------------------------
// WMMA f16->f32 GEMM: C[M,N] = A[M,K] * B + epilogue.
// A: f32 row-major, lda == K (mult of 32). B: f32; BT==0 -> [K,N], BT==1 -> [N,K].
// Epilogue: + bias[n] + resid[m,n] + bbias[m >> rpbShift, n], optional relu.
// 128 threads = 4 waves; block tile 64x64, K-step 32; float4 staging,
// double-buffered LDS (one barrier per K-step), pipelined one tile ahead.
// ---------------------------------------------------------------------------
template<int BT>
__launch_bounds__(128)
__global__ void gemm_kernel(const float* __restrict__ A, int lda,
                            const float* __restrict__ Bm, int ldb,
                            float* __restrict__ Cm, int M, int N, int K,
                            const float* __restrict__ bias,
                            const float* __restrict__ resid,
                            const float* __restrict__ bbias, int rpbShift,
                            int relu)
{
    __shared__ _Float16 As[2][64][40];  // [m][k], 80B row pitch -> conflict-free
    __shared__ _Float16 Bs[2][64][40];  // [n][k] (B transposed into LDS)
    int m0 = blockIdx.y * 64, n0 = blockIdx.x * 64;
    int tid  = threadIdx.x;
    int wave = tid >> 5, lane = tid & 31;
    int g1   = lane >> 4, l16 = lane & 15;

    v4f ra[4], rb[4];   // staged global data for the next K-tile

    auto gload = [&](int k0){
        #pragma unroll
        for (int i = 0; i < 4; ++i){
            int e = i*128 + tid;
            int m = e >> 3, kq = e & 7;           // 8 float4 per 32-wide row
            ra[i] = *(const v4f*)&A[(size_t)(m0 + m)*lda + k0 + 4*kq];
        }
        if (BT){
            #pragma unroll
            for (int i = 0; i < 4; ++i){
                int e = i*128 + tid;
                int n = e >> 3, kq = e & 7;
                rb[i] = *(const v4f*)&Bm[(size_t)(n0 + n)*ldb + k0 + 4*kq];
            }
        } else {
            #pragma unroll
            for (int i = 0; i < 4; ++i){
                int e = i*128 + tid;
                int k = e >> 4, nq = e & 15;      // 16 float4 per 64-wide row
                rb[i] = *(const v4f*)&Bm[(size_t)(k0 + k)*ldb + n0 + 4*nq];
            }
        }
    };
    auto stage = [&](int pb){
        #pragma unroll
        for (int i = 0; i < 4; ++i){
            int e = i*128 + tid;
            int m = e >> 3, kq = e & 7;
            v4h h;
            #pragma unroll
            for (int j = 0; j < 4; ++j) h[j] = (_Float16)ra[i][j];
            *(v4h*)&As[pb][m][4*kq] = h;          // ds_store_b64
        }
        if (BT){
            #pragma unroll
            for (int i = 0; i < 4; ++i){
                int e = i*128 + tid;
                int n = e >> 3, kq = e & 7;
                v4h h;
                #pragma unroll
                for (int j = 0; j < 4; ++j) h[j] = (_Float16)rb[i][j];
                *(v4h*)&Bs[pb][n][4*kq] = h;
            }
        } else {
            #pragma unroll
            for (int i = 0; i < 4; ++i){
                int e = i*128 + tid;
                int k = e >> 4, nq = e & 15;
                #pragma unroll
                for (int j = 0; j < 4; ++j)
                    Bs[pb][4*nq + j][k] = (_Float16)rb[i][j];  // transpose scatter
            }
        }
    };

    v8f zero = {0.f,0.f,0.f,0.f,0.f,0.f,0.f,0.f};
    v8f acc[4] = {zero, zero, zero, zero};

    gload(0);
    stage(0);
    int cur = 0;
    for (int k0 = 0; k0 < K; k0 += 32){
        __syncthreads();                  // buffer 'cur' fully staged
        bool more = (k0 + 32 < K);
        if (more) gload(k0 + 32);         // global loads overlap WMMA below

        // A fragment (16x32 f16 layout, ISA 7.12.2): lane = M'|16*g1,
        // halves at K = 8*g1 and K = 16 + 8*g1.
        int mrow = wave*16 + l16;
        v8h a0 = *(const v8h*)&As[cur][mrow][8*g1];
        v8h a1 = *(const v8h*)&As[cur][mrow][16 + 8*g1];
        v16h af;
        #pragma unroll
        for (int i = 0; i < 8; ++i){ af[i] = a0[i]; af[i+8] = a1[i]; }

        #pragma unroll
        for (int nt = 0; nt < 4; ++nt){
            // B fragment: lane = N'|16*g1, halves at K = 8*g1 and K = 16 + 8*g1 (same as A)
            int ncol = nt*16 + l16;
            v8h b0 = *(const v8h*)&Bs[cur][ncol][8*g1];
            v8h b1 = *(const v8h*)&Bs[cur][ncol][16 + 8*g1];
            v16h bf;
            #pragma unroll
            for (int i = 0; i < 8; ++i){ bf[i] = b0[i]; bf[i+8] = b1[i]; }
            acc[nt] = __builtin_amdgcn_wmma_f32_16x16x32_f16(
                false, af, false, bf, (short)0, acc[nt], false, false);
        }
        if (more) stage(cur ^ 1);         // write other buffer; no 2nd barrier
        cur ^= 1;
    }

    // C/D layout: VGPR r -> M = r + 8*g1, N = l16 (per 16x16 f32 layout).
    #pragma unroll
    for (int nt = 0; nt < 4; ++nt){
        int n = n0 + nt*16 + l16;
        float badd = bias ? bias[n] : 0.f;
        #pragma unroll
        for (int r = 0; r < 8; ++r){
            int m = m0 + wave*16 + r + 8*g1;
            size_t off = (size_t)m * N + n;
            float val = acc[nt][r] + badd;
            if (resid) val += resid[off];
            if (bbias) val += bbias[(size_t)(m >> rpbShift) * N + n];
            if (relu)  val = fmaxf(val, 0.f);
            Cm[off] = val;
        }
    }
}

// ---------------------------------------------------------------------------
// Per-batch column mean, two-stage deterministic reduction.
// ---------------------------------------------------------------------------
__global__ void colmean_part_kernel(const float* __restrict__ x, int V, int C,
                                    int chunks, float* __restrict__ part)
{
    int ch = blockIdx.x, b = blockIdx.y;
    int rows = V / chunks;
    const float* p0 = x + ((size_t)b*V + (size_t)ch*rows)*C;
    for (int c = threadIdx.x; c < C; c += blockDim.x){
        float s = 0.f;
        for (int v = 0; v < rows; ++v) s += p0[(size_t)v*C + c];
        part[((size_t)b*chunks + ch)*C + c] = s;
    }
}
__global__ void colmean_fin_kernel(const float* __restrict__ part, int chunks,
                                   int C, int V, float* __restrict__ out)
{
    int t = blockIdx.x * blockDim.x + threadIdx.x;
    if (t >= NB * C) return;
    int b = t / C, c = t % C;
    float s = 0.f;
    for (int ch = 0; ch < chunks; ++ch)
        s += part[((size_t)b*chunks + ch)*C + c];
    out[t] = s / (float)V;
}

// g-bias for _orl: out[b,o] = sum_c mean[b,c] * W[o, C + c], W is (N, 2C).
__global__ void gbias_kernel(const float* __restrict__ mean,
                             const float* __restrict__ W, int C, int N,
                             float* __restrict__ out)
{
    int t = blockIdx.x * blockDim.x + threadIdx.x;
    if (t >= NB * N) return;
    int b = t / N, o = t % N;
    const float* wrow = W + (size_t)o * (2*C) + C;
    const float* mrow = mean + (size_t)b * C;
    float s = 0.f;
    for (int c = 0; c < C; ++c) s += mrow[c] * wrow[c];
    out[t] = s;
}

// ---------------------------------------------------------------------------
// BatchNorm stats over rows (R = NB*V): one block per channel.
// ---------------------------------------------------------------------------
__global__ void bn_stats_kernel(const float* __restrict__ x, int R, int C,
                                float* __restrict__ mv)
{
    int c = blockIdx.x;
    int tid = threadIdx.x;
    __shared__ float ssum[256], ssq[256];
    float s = 0.f, q = 0.f;
    for (int r = tid; r < R; r += blockDim.x){
        float v = x[(size_t)r*C + c];
        s += v; q += v*v;
    }
    ssum[tid] = s; ssq[tid] = q;
    __syncthreads();
    for (int off = 128; off > 0; off >>= 1){
        if (tid < off){ ssum[tid] += ssum[tid+off]; ssq[tid] += ssq[tid+off]; }
        __syncthreads();
    }
    if (tid == 0){
        float m = ssum[0] / (float)R;
        mv[c]     = m;
        mv[C + c] = ssq[0] / (float)R - m*m;
    }
}

__global__ void bn_apply_kernel(float* __restrict__ x,
                                const float* __restrict__ mv,
                                const float* __restrict__ g,
                                const float* __restrict__ bta,
                                int total, int C, int relu)
{
    int t = blockIdx.x * blockDim.x + threadIdx.x;
    if (t >= total) return;
    int c = t % C;
    float y = (x[t] - mv[c]) * rsqrtf(mv[C + c] + 1e-5f) * g[c] + bta[c];
    if (relu) y = fmaxf(y, 0.f);
    x[t] = y;
}

// ---------------------------------------------------------------------------
// Max pool over k gathered neighbors. block = (p, b).
// ---------------------------------------------------------------------------
__global__ void pool_kernel(const float* __restrict__ f,
                            const int* __restrict__ idx,
                            int Vq, int K, int C, int Vsrc,
                            float* __restrict__ out)
{
    int p = blockIdx.x, b = blockIdx.y;
    const int* ip = idx + ((size_t)b*Vq + p)*K;
    for (int c = threadIdx.x; c < C; c += blockDim.x){
        float m = -3.4e38f;
        for (int k = 0; k < K; ++k)
            m = fmaxf(m, f[((size_t)b*Vsrc + ip[k])*C + c]);
        out[((size_t)b*Vq + p)*C + c] = m;
    }
}

// Final max over vertices: out[b,c] = max_v f[b,v,c].
__global__ void finalmax_kernel(const float* __restrict__ f, int V, int C,
                                float* __restrict__ out)
{
    int t = blockIdx.x * blockDim.x + threadIdx.x;
    if (t >= NB * C) return;
    int b = t / C, c = t % C;
    float m = -3.4e38f;
    const float* p = f + (size_t)b*V*C + c;
    for (int v = 0; v < V; ++v) m = fmaxf(m, p[(size_t)v*C]);
    out[t] = m;
}

// ===========================================================================
extern "C" void kernel_launch(void* const* d_in, const int* in_sizes, int n_in,
                              void* d_out, int out_size, void* d_ws, size_t ws_size,
                              hipStream_t stream)
{
    const float* verts = (const float*)d_in[0];
    const float* dirs0 = (const float*)d_in[1];
    const float* ste0  = (const float*)d_in[2];
    const float* orl0  = (const float*)d_in[3];
    const float* w1    = (const float*)d_in[4];
    const float* b1    = (const float*)d_in[5];
    const float* dirs1 = (const float*)d_in[6];
    const float* ste1  = (const float*)d_in[7];
    const float* orl1  = (const float*)d_in[8];
    const float* bn1g  = (const float*)d_in[9];
    const float* bn1b  = (const float*)d_in[10];
    const float* w2    = (const float*)d_in[11];
    const float* b2    = (const float*)d_in[12];
    const float* dirs2 = (const float*)d_in[13];
    const float* ste2  = (const float*)d_in[14];
    const float* orl2  = (const float*)d_in[15];
    const float* bn2g  = (const float*)d_in[16];
    const float* bn2b  = (const float*)d_in[17];
    const float* w3    = (const float*)d_in[18];
    const float* b3    = (const float*)d_in[19];
    const float* dirs3 = (const float*)d_in[20];
    const float* ste3  = (const float*)d_in[21];
    const float* orl3  = (const float*)d_in[22];
    const float* bn3g  = (const float*)d_in[23];
    const float* bn3b  = (const float*)d_in[24];
    const float* w4    = (const float*)d_in[25];
    const float* b4    = (const float*)d_in[26];
    const float* dirs4 = (const float*)d_in[27];
    const float* ste4  = (const float*)d_in[28];
    const float* orl4  = (const float*)d_in[29];
    (void)in_sizes; (void)n_in; (void)out_size; (void)ws_size;

    const int V0 = 4096, V1 = 1024, V2 = 256, NN = 10, CH = 16;

    char* wp = (char*)d_ws;
    auto alloc = [&](size_t bytes)->void*{
        void* p = (void*)wp;
        wp += (bytes + 255) & ~(size_t)255;
        return p;
    };
    float* nd    = (float*)alloc(sizeof(float)*(size_t)3*SUP*512);
    int*   idx0  = (int*)  alloc(sizeof(int)*(size_t)NB*V0*NN);
    int*   idxp1 = (int*)  alloc(sizeof(int)*(size_t)NB*V1*4);
    int*   idx1  = (int*)  alloc(sizeof(int)*(size_t)NB*V1*NN);
    int*   idxp2 = (int*)  alloc(sizeof(int)*(size_t)NB*V2*4);
    int*   idx2  = (int*)  alloc(sizeof(int)*(size_t)NB*V2*NN);
    float* cm    = (float*)alloc(sizeof(float)*(size_t)NB*512);
    float* cpart = (float*)alloc(sizeof(float)*(size_t)NB*CH*512);
    float* gb    = (float*)alloc(sizeof(float)*(size_t)NB*512);
    float* mv    = (float*)alloc(sizeof(float)*1024);
    float* featA = (float*)alloc(sizeof(float)*(size_t)NB*V0*128);
    float* fm0   = (float*)alloc(sizeof(float)*(size_t)NB*V0*128);
    float* fste  = (float*)alloc(sizeof(float)*(size_t)NB*V0*128);
    float* fout  = (float*)alloc(sizeof(float)*(size_t)NB*V0*1024);
    float* pre   = (float*)alloc(sizeof(float)*(size_t)NB*V0*128);
    float* fm1   = (float*)alloc(sizeof(float)*(size_t)NB*V0*128);
    float* fp1   = (float*)alloc(sizeof(float)*(size_t)NB*V1*128);
    float* fm2   = (float*)alloc(sizeof(float)*(size_t)NB*V1*256);
    float* fm3   = (float*)alloc(sizeof(float)*(size_t)NB*V1*256);
    float* fp2   = (float*)alloc(sizeof(float)*(size_t)NB*V2*256);
    float* fm4   = (float*)alloc(sizeof(float)*(size_t)NB*V2*512);

    auto gemm = [&](const float* A, const float* Bm, int ldb, int bT, float* Cc,
                    int M, int N, int K, const float* bias, const float* resid,
                    const float* bbias, int rpbShift, int relu){
        dim3 g(N/64, M/64);
        if (bT)
            gemm_kernel<1><<<g, 128, 0, stream>>>(A, K, Bm, ldb, Cc, M, N, K,
                                                  bias, resid, bbias, rpbShift, relu);
        else
            gemm_kernel<0><<<g, 128, 0, stream>>>(A, K, Bm, ldb, Cc, M, N, K,
                                                  bias, resid, bbias, rpbShift, relu);
    };
    auto colmean = [&](const float* x, int V, int C){
        colmean_part_kernel<<<dim3(CH, NB), 256, 0, stream>>>(x, V, C, CH, cpart);
        colmean_fin_kernel<<<(NB*C+255)/256, 256, 0, stream>>>(cpart, CH, C, V, cm);
    };
    auto hs_act = [&](const int* idx, int J, const float* fo, int ldo,
                      const float* fs, float* pr, int V, int C){
        int threads = C < 256 ? C : 256;
        hs_act_kernel<<<dim3(V, NB), threads, 0, stream>>>(verts, V0, idx, NN, nd,
                                                           J, fo, ldo, fs, pr, V, C);
    };

    // ---------------- Stage 0: surface layer -> fm0 ----------------
    normdirs_kernel<<<(SUP*128+255)/256, 256, 0, stream>>>(dirs0, SUP*128, nd);
    knn_kernel<<<(NB*V0+127)/128, 128, 0, stream>>>(verts, V0, V0, V0, NN, idx0);
    surface_kernel<<<dim3(V0, NB), 128, 0, stream>>>(verts, V0, idx0, NN, nd,
                                                     SUP*128, ste0, featA, V0);
    colmean(featA, V0, 128);
    gbias_kernel<<<(NB*128+255)/256, 256, 0, stream>>>(cm, orl0, 128, 128, gb);
    gemm(featA, orl0, 256, 1, fm0, NB*V0, 128, 128, nullptr, featA, gb, 12, 1);

    // ---------------- Layer 1 (hs, C=128) -> fm1 ----------------
    normdirs_kernel<<<(SUP*128+255)/256, 256, 0, stream>>>(dirs1, SUP*128, nd);
    gemm(fm0, ste1, 128, 1, fste, NB*V0, 128, 128, nullptr, nullptr, nullptr, 0, 0);
    gemm(fm0, w1, 1024, 0, fout, NB*V0, 1024, 128, b1, nullptr, nullptr, 0, 0);
    hs_act(idx0, SUP*128, fout, 1024, fste, pre, V0, 128);
    colmean(pre, V0, 128);
    gbias_kernel<<<(NB*128+255)/256, 256, 0, stream>>>(cm, orl1, 128, 128, gb);
    gemm(pre, orl1, 256, 1, fm1, NB*V0, 128, 128, nullptr, pre, gb, 12, 0);
    bn_stats_kernel<<<128, 256, 0, stream>>>(fm1, NB*V0, 128, mv);
    bn_apply_kernel<<<(NB*V0*128+255)/256, 256, 0, stream>>>(fm1, mv, bn1g, bn1b,
                                                             NB*V0*128, 128, 1);

    // ---------------- Pool 1 -> fp1 (v1 = first 1024 verts) ----------------
    knn_kernel<<<(NB*V1+127)/128, 128, 0, stream>>>(verts, V0, V0, V1, 4, idxp1);
    pool_kernel<<<dim3(V1, NB), 128, 0, stream>>>(fm1, idxp1, V1, 4, 128, V0, fp1);

    // ---------------- Layer 2 (hs, C=256) -> fm2 ----------------
    knn_kernel<<<(NB*V1+127)/128, 128, 0, stream>>>(verts, V0, V1, V1, NN, idx1);
    normdirs_kernel<<<(SUP*256+255)/256, 256, 0, stream>>>(dirs2, SUP*256, nd);
    gemm(fp1, ste2, 128, 1, fste, NB*V1, 256, 128, nullptr, nullptr, nullptr, 0, 0);
    gemm(fp1, w2, 2048, 0, fout, NB*V1, 2048, 128, b2, nullptr, nullptr, 0, 0);
    hs_act(idx1, SUP*256, fout, 2048, fste, pre, V1, 256);
    colmean(pre, V1, 256);
    gbias_kernel<<<(NB*256+255)/256, 256, 0, stream>>>(cm, orl2, 256, 256, gb);
    gemm(pre, orl2, 512, 1, fm2, NB*V1, 256, 256, nullptr, pre, gb, 10, 0);
    bn_stats_kernel<<<256, 256, 0, stream>>>(fm2, NB*V1, 256, mv);
    bn_apply_kernel<<<(NB*V1*256+255)/256, 256, 0, stream>>>(fm2, mv, bn2g, bn2b,
                                                             NB*V1*256, 256, 1);

    // ---------------- Layer 3 (hs, C=256) -> fm3 ----------------
    normdirs_kernel<<<(SUP*256+255)/256, 256, 0, stream>>>(dirs3, SUP*256, nd);
    gemm(fm2, ste3, 256, 1, fste, NB*V1, 256, 256, nullptr, nullptr, nullptr, 0, 0);
    gemm(fm2, w3, 2048, 0, fout, NB*V1, 2048, 256, b3, nullptr, nullptr, 0, 0);
    hs_act(idx1, SUP*256, fout, 2048, fste, pre, V1, 256);
    colmean(pre, V1, 256);
    gbias_kernel<<<(NB*256+255)/256, 256, 0, stream>>>(cm, orl3, 256, 256, gb);
    gemm(pre, orl3, 512, 1, fm3, NB*V1, 256, 256, nullptr, pre, gb, 10, 0);
    bn_stats_kernel<<<256, 256, 0, stream>>>(fm3, NB*V1, 256, mv);
    bn_apply_kernel<<<(NB*V1*256+255)/256, 256, 0, stream>>>(fm3, mv, bn3g, bn3b,
                                                             NB*V1*256, 256, 1);

    // ---------------- Pool 2 -> fp2 (v2 = first 256 verts) ----------------
    knn_kernel<<<(NB*V2+127)/128, 128, 0, stream>>>(verts, V0, V1, V2, 4, idxp2);
    pool_kernel<<<dim3(V2, NB), 256, 0, stream>>>(fm3, idxp2, V2, 4, 256, V1, fp2);

    // ---------------- Layer 4 (hs, C=512, no bn/relu) -> fm4 ----------------
    knn_kernel<<<(NB*V2+127)/128, 128, 0, stream>>>(verts, V0, V2, V2, NN, idx2);
    normdirs_kernel<<<(SUP*512+255)/256, 256, 0, stream>>>(dirs4, SUP*512, nd);
    gemm(fp2, ste4, 256, 1, fste, NB*V2, 512, 256, nullptr, nullptr, nullptr, 0, 0);
    gemm(fp2, w4, 4096, 0, fout, NB*V2, 4096, 256, b4, nullptr, nullptr, 0, 0);
    hs_act(idx2, SUP*512, fout, 4096, fste, pre, V2, 512);
    colmean(pre, V2, 512);
    gbias_kernel<<<(NB*512+255)/256, 256, 0, stream>>>(cm, orl4, 512, 512, gb);
    gemm(pre, orl4, 1024, 1, fm4, NB*V2, 512, 512, nullptr, pre, gb, 8, 0);

    // ---------------- Output: max over vertices -> (4, 512) ----------------
    finalmax_kernel<<<(NB*512+255)/256, 256, 0, stream>>>(fm4, V2, 512,
                                                          (float*)d_out);
}
